// LSTM_27711128993879
// MI455X (gfx1250) — matrix operand order repaired and checked by reading, not compiled, not run on hardware
//
#include <hip/hip_runtime.h>
#include <hip/hip_bf16.h>
#include <stdint.h>

// ---------------------------------------------------------------------------
// Types for CDNA5 WMMA (gfx1250, wave32)
// ---------------------------------------------------------------------------
typedef __bf16 bf16_t;
typedef __attribute__((ext_vector_type(16))) __bf16 v16bf;
typedef __attribute__((ext_vector_type(8)))  __bf16 v8bf;
typedef __attribute__((ext_vector_type(8)))  float  v8f;

union Frag16 { v16bf v; v8bf h[2]; };

static __device__ __forceinline__ bf16_t f2bf(float f) {
    uint32_t u = __builtin_bit_cast(uint32_t, f);
    uint32_t r = u + 0x7FFFu + ((u >> 16) & 1u);   // round-to-nearest-even
    uint16_t s = (uint16_t)(r >> 16);
    return __builtin_bit_cast(bf16_t, s);
}
static __device__ __forceinline__ float bf2f(bf16_t b) {
    uint16_t s = __builtin_bit_cast(uint16_t, b);
    uint32_t u = (uint32_t)s << 16;
    return __builtin_bit_cast(float, u);
}

static __device__ __forceinline__ float sigmoid_(float x) {
    return 1.0f / (1.0f + __expf(-x));
}
static __device__ __forceinline__ float tanh_(float x) {
    float ax = fabsf(x);
    float e  = __expf(-2.0f * ax);
    float t  = (1.0f - e) / (1.0f + e);
    return copysignf(t, x);
}

#define WMMA_BF16(A, B, C) \
    __builtin_amdgcn_wmma_f32_16x16x32_bf16(false, (A), false, (B), (short)0, (C), false, false)

// Problem constants
#define BATCH 64
#define TSTEPS 2048
#define DIMS 256
#define HID 512
#define GCOLS 2048   // 4 gates * HID

// ---------------------------------------------------------------------------
// Conversion kernels: build bf16, K-contiguous (transposed) weight copies
// ---------------------------------------------------------------------------
__global__ void k_f32_to_bf16(const float* __restrict__ src, bf16_t* __restrict__ dst, long n) {
    for (long i = (long)blockIdx.x * blockDim.x + threadIdx.x; i < n;
         i += (long)gridDim.x * blockDim.x)
        dst[i] = f2bf(src[i]);
}

// WcatT[col][d], col = g*512 + h  (2048 x 256)
__global__ void k_cvt_wcat(const float* __restrict__ Wj, const float* __restrict__ Wi,
                           const float* __restrict__ Wf, const float* __restrict__ Wo,
                           bf16_t* __restrict__ WcatT) {
    long i = (long)blockIdx.x * blockDim.x + threadIdx.x;
    if (i >= (long)GCOLS * DIMS) return;
    int col = (int)(i >> 8);        // / 256
    int d   = (int)(i & 255);
    int g   = col >> 9;
    int h   = col & 511;
    const float* W = (g == 0) ? Wj : (g == 1) ? Wi : (g == 2) ? Wf : Wo;
    WcatT[i] = f2bf(W[(size_t)d * HID + h]);
}

// UcatT[col][k], col = g*512 + h  (2048 x 512)
__global__ void k_cvt_ucat(const float* __restrict__ Uj, const float* __restrict__ Ui,
                           const float* __restrict__ Uf, const float* __restrict__ Uo,
                           bf16_t* __restrict__ UcatT) {
    long i = (long)blockIdx.x * blockDim.x + threadIdx.x;
    if (i >= (long)GCOLS * HID) return;
    int col = (int)(i >> 9);        // / 512
    int k   = (int)(i & 511);
    int g   = col >> 9;
    int h   = col & 511;
    const float* U = (g == 0) ? Uj : (g == 1) ? Ui : (g == 2) ? Uf : Uo;
    UcatT[i] = f2bf(U[(size_t)k * HID + h]);
}

// WyT[dcol][k]  (256 x 512)
__global__ void k_cvt_wy(const float* __restrict__ Wy, bf16_t* __restrict__ WyT) {
    long i = (long)blockIdx.x * blockDim.x + threadIdx.x;
    if (i >= (long)DIMS * HID) return;
    int dcol = (int)(i >> 9);
    int k    = (int)(i & 511);
    WyT[i] = f2bf(Wy[(size_t)k * DIMS + dcol]);
}

// ---------------------------------------------------------------------------
// Phase 1: input projections.  pre[t][b][col] = x[b][t][:] . Wcat[:,col] + bias
// Stored as bf16 (halves HBM traffic; matches bf16 pipeline accuracy).
// Each wave computes a 16(M: time) x 64(N: 4 tiles) output block.
// ---------------------------------------------------------------------------
__global__ void k_input_proj(const bf16_t* __restrict__ xbf, const bf16_t* __restrict__ WcatT,
                             const float* __restrict__ bj, const float* __restrict__ bi,
                             const float* __restrict__ bf_, const float* __restrict__ bo,
                             bf16_t* __restrict__ pre) {
    int wv   = threadIdx.x >> 5;
    int lane = threadIdx.x & 31;
    int ln   = lane & 15;
    int hi   = lane >> 4;

    long job = (long)blockIdx.x * 8 + wv;   // 262144 wave jobs
    int mt = (int)(job >> 5);               // 0..8191  (M tiles over (b,t))
    int ng = (int)(job & 31);               // 0..31    (64-col groups)
    int b  = mt >> 7;
    int t0 = (mt & 127) << 4;
    int n0 = ng << 6;

    const bf16_t* arow = xbf + ((size_t)b * TSTEPS + t0 + ln) * DIMS;

    v8f acc[4] = {};
    for (int k0 = 0; k0 < DIMS; k0 += 32) {
        Frag16 A;
        A.h[0] = *(const v8bf*)(arow + k0 + hi * 8);
        A.h[1] = *(const v8bf*)(arow + k0 + 16 + hi * 8);
#pragma unroll
        for (int j = 0; j < 4; ++j) {
            Frag16 B;
            const bf16_t* brow = WcatT + (size_t)(n0 + j * 16 + ln) * DIMS + hi * 8 + k0;
            B.h[0] = *(const v8bf*)(brow);
            B.h[1] = *(const v8bf*)(brow + 16);
            acc[j] = WMMA_BF16(A.v, B.v, acc[j]);
        }
    }

#pragma unroll
    for (int j = 0; j < 4; ++j) {
        int col  = n0 + j * 16 + ln;
        int g    = col >> 9;
        int hcol = col & 511;
        const float* bias = (g == 0) ? bj : (g == 1) ? bi : (g == 2) ? bf_ : bo;
        float bv = bias[hcol];
#pragma unroll
        for (int v = 0; v < 8; ++v) {
            int M = v + hi * 8;
            pre[((size_t)(t0 + M) * BATCH + b) * GCOLS + col] = f2bf(acc[j][v] + bv);
        }
    }
}

// ---------------------------------------------------------------------------
// Phase 2: recurrent scan.  4 persistent workgroups, each owns 16 batch rows.
// Wave w owns hidden columns [w*16, w*16+16) for ALL four gates -> gate fusion
// and running c entirely in registers; h exchanged through LDS each step.
// Pre-activation tiles are double-buffered in LDS via async global->LDS DMA
// (GLOBAL_LOAD_ASYNC_TO_LDS_B128, ASYNCcnt), overlapping HBM latency with the
// recurrent WMMA GEMM. Each wave copies exactly the columns it later reads,
// so synchronization is per-wave s_wait_asynccnt only.
// ---------------------------------------------------------------------------
__device__ __forceinline__ void issue_pre_async(const bf16_t* __restrict__ pre,
                                                bf16_t* plds_base, int t, int buf,
                                                int b0, int wv, int lane) {
#pragma unroll
    for (int i = 0; i < 4; ++i) {
        int id = lane + 32 * i;          // 0..127 chunks per wave
        int g  = id >> 5;                // gate
        int M  = (id >> 1) & 15;         // row
        int cq = id & 1;                 // 16B chunk within the 16-col slice
        int col = g * HID + wv * 16 + cq * 8;
        const bf16_t* gp = pre + ((size_t)t * BATCH + b0 + M) * GCOLS + col;
        bf16_t* lp = plds_base + ((size_t)buf * 16 + M) * GCOLS + col;
        uint32_t lds_off = (uint32_t)(size_t)lp;     // low 32 bits = LDS byte offset
        asm volatile("global_load_async_to_lds_b128 %0, %1, off"
                     :: "v"(lds_off), "v"((uint64_t)(size_t)gp)
                     : "memory");
    }
}

__global__ __launch_bounds__(1024, 1)
void k_scan(const bf16_t* __restrict__ pre, const bf16_t* __restrict__ UcatT,
            const float* __restrict__ h0, const float* __restrict__ c0,
            bf16_t* __restrict__ hs, float* __restrict__ out_hlast,
            float* __restrict__ out_clast) {
    __shared__ __align__(16) bf16_t hsh[16 * HID];          // 16 KB: h slice
    __shared__ __align__(16) bf16_t plds[2][16 * GCOLS];    // 2 x 64 KB: pre tiles

    int tid  = threadIdx.x;
    int wv   = tid >> 5;
    int lane = tid & 31;
    int ln   = lane & 15;
    int hi   = lane >> 4;
    int b0   = blockIdx.x * 16;
    int mycol = wv * 16 + ln;                        // owned hidden column

    // kick off async copy of the t=0 pre tile while we initialize
    issue_pre_async(pre, &plds[0][0], 0, 0, b0, wv, lane);

    // init LDS h from h0 (f32 [64][512])
    for (int i = tid; i < 16 * HID; i += 1024) {
        int r = i >> 9, cc = i & 511;
        hsh[i] = f2bf(h0[(size_t)(b0 + r) * HID + cc]);
    }
    // init running c (C-tile layout: VGPR v holds row M = v + hi*8, col = mycol)
    float creg[8], hcur[8];
#pragma unroll
    for (int v = 0; v < 8; ++v) {
        creg[v] = c0[(size_t)(b0 + v + hi * 8) * HID + mycol];
        hcur[v] = 0.0f;
    }
    const bf16_t* Ub[4];
#pragma unroll
    for (int g = 0; g < 4; ++g)
        Ub[g] = UcatT + (size_t)(g * HID + mycol) * HID + hi * 8;

    __syncthreads();

    for (int t = 0; t < TSTEPS; ++t) {
        // issue async copy for next step's tile (clamped dummy on last step so
        // the ASYNCcnt discipline stays uniform: outstanding is always 8 here)
        int tn = (t + 1 < TSTEPS) ? (t + 1) : (TSTEPS - 1);
        issue_pre_async(pre, &plds[0][0], tn, (t + 1) & 1, b0, wv, lane);

        // recurrent GEMM: acc[g] += h(16x512) . U_g(512x16) for owned columns
        v8f acc[4] = {};
        for (int k0 = 0; k0 < HID; k0 += 32) {
            Frag16 A;
            const bf16_t* ar = &hsh[(size_t)ln * HID + k0 + hi * 8];
            A.h[0] = *(const v8bf*)(ar);
            A.h[1] = *(const v8bf*)(ar + 16);
#pragma unroll
            for (int g = 0; g < 4; ++g) {
                Frag16 B;
                B.h[0] = *(const v8bf*)(Ub[g] + k0);
                B.h[1] = *(const v8bf*)(Ub[g] + k0 + 16);
                acc[g] = WMMA_BF16(A.v, B.v, acc[g]);
            }
        }

        // wait until this step's async tile has landed (t+1's 4 may remain)
        asm volatile("s_wait_asynccnt 0x4" ::: "memory");

        const bf16_t* pt = &plds[t & 1][0];
#pragma unroll
        for (int v = 0; v < 8; ++v) {
            int M = v + hi * 8;
            const bf16_t* rb = pt + (size_t)M * GCOLS;
            float pj = bf2f(rb[0 * HID + mycol]);
            float pi = bf2f(rb[1 * HID + mycol]);
            float pf = bf2f(rb[2 * HID + mycol]);
            float po = bf2f(rb[3 * HID + mycol]);
            // reference: j,i,o sigmoid; f tanh; cell tanh
            float gj = sigmoid_(pj + acc[0][v]);
            float gi = sigmoid_(pi + acc[1][v]);
            float gf = tanh_(pf + acc[2][v]);
            float go = sigmoid_(po + acc[3][v]);
            float cc = gf * creg[v] + gi * gj;
            creg[v]  = cc;
            hcur[v]  = go * tanh_(cc);
        }

        __syncthreads();   // all waves finished reading old h from LDS
#pragma unroll
        for (int v = 0; v < 8; ++v) {
            int M = v + hi * 8;
            bf16_t hb = f2bf(hcur[v]);
            hsh[(size_t)M * HID + mycol] = hb;
            hs[((size_t)t * BATCH + b0 + M) * HID + mycol] = hb;
        }
        __syncthreads();   // new h visible before next step's fragment loads
    }

#pragma unroll
    for (int v = 0; v < 8; ++v) {
        int M = v + hi * 8;
        out_hlast[(size_t)(b0 + M) * HID + mycol] = hcur[v];
        out_clast[(size_t)(b0 + M) * HID + mycol] = creg[v];
    }
}

// ---------------------------------------------------------------------------
// Phase 3: output projection.  y[b][t][d] = hs[t][b][:] . Wy[:,d] + b_y
// Each wave: 16(M: batch) x 64(N) block at fixed t.
// ---------------------------------------------------------------------------
__global__ void k_out_proj(const bf16_t* __restrict__ hs, const bf16_t* __restrict__ WyT,
                           const float* __restrict__ by, float* __restrict__ y) {
    int wv   = threadIdx.x >> 5;
    int lane = threadIdx.x & 31;
    int ln   = lane & 15;
    int hi   = lane >> 4;

    long job = (long)blockIdx.x * 8 + wv;   // 32768 wave jobs
    int mt = (int)(job >> 2);               // 0..8191
    int ng = (int)(job & 3);
    int t  = mt >> 2;
    int b0 = (mt & 3) << 4;
    int n0 = ng << 6;

    const bf16_t* arow = hs + ((size_t)t * BATCH + b0 + ln) * HID;

    v8f acc[4] = {};
    for (int k0 = 0; k0 < HID; k0 += 32) {
        Frag16 A;
        A.h[0] = *(const v8bf*)(arow + k0 + hi * 8);
        A.h[1] = *(const v8bf*)(arow + k0 + 16 + hi * 8);
#pragma unroll
        for (int j = 0; j < 4; ++j) {
            Frag16 B;
            const bf16_t* brow = WyT + (size_t)(n0 + j * 16 + ln) * HID + hi * 8 + k0;
            B.h[0] = *(const v8bf*)(brow);
            B.h[1] = *(const v8bf*)(brow + 16);
            acc[j] = WMMA_BF16(A.v, B.v, acc[j]);
        }
    }

#pragma unroll
    for (int j = 0; j < 4; ++j) {
        int d  = n0 + j * 16 + ln;
        float bv = by[d];
#pragma unroll
        for (int v = 0; v < 8; ++v) {
            int M = v + hi * 8;
            y[((size_t)(b0 + M) * TSTEPS + t) * DIMS + d] = acc[j][v] + bv;
        }
    }
}

// ---------------------------------------------------------------------------
// Host launcher
// ---------------------------------------------------------------------------
extern "C" void kernel_launch(void* const* d_in, const int* in_sizes, int n_in,
                              void* d_out, int out_size, void* d_ws, size_t ws_size,
                              hipStream_t stream) {
    const float* x   = (const float*)d_in[0];
    const float* h0  = (const float*)d_in[1];
    const float* c0  = (const float*)d_in[2];
    const float* Wj  = (const float*)d_in[3];
    const float* Wi  = (const float*)d_in[4];
    const float* Wf  = (const float*)d_in[5];
    const float* Wo  = (const float*)d_in[6];
    const float* Uj  = (const float*)d_in[7];
    const float* Ui  = (const float*)d_in[8];
    const float* Uf  = (const float*)d_in[9];
    const float* Uo  = (const float*)d_in[10];
    const float* bj  = (const float*)d_in[11];
    const float* bi  = (const float*)d_in[12];
    const float* bf_ = (const float*)d_in[13];
    const float* bo  = (const float*)d_in[14];
    const float* Wy  = (const float*)d_in[15];
    const float* by  = (const float*)d_in[16];

    // Workspace layout (bf16 element counts)
    uint8_t* ws = (uint8_t*)d_ws;
    bf16_t* xbf   = (bf16_t*)(ws);                         // 64*2048*256
    bf16_t* WcatT = xbf + (size_t)BATCH * TSTEPS * DIMS;   // 2048*256
    bf16_t* UcatT = WcatT + (size_t)GCOLS * DIMS;          // 2048*512
    bf16_t* WyT   = UcatT + (size_t)GCOLS * HID;           // 256*512
    bf16_t* hs    = WyT + (size_t)DIMS * HID;              // 2048*64*512
    bf16_t* pre   = hs + (size_t)TSTEPS * BATCH * HID;     // 2048*64*2048 (bf16)

    float* y      = (float*)d_out;
    float* hlast  = y + (size_t)BATCH * TSTEPS * DIMS;
    float* clast  = hlast + (size_t)BATCH * HID;

    // Phase 0: conversions
    long nx = (long)BATCH * TSTEPS * DIMS;
    k_f32_to_bf16<<<4096, 256, 0, stream>>>(x, xbf, nx);
    k_cvt_wcat<<<(GCOLS * DIMS + 255) / 256, 256, 0, stream>>>(Wj, Wi, Wf, Wo, WcatT);
    k_cvt_ucat<<<(GCOLS * HID + 255) / 256, 256, 0, stream>>>(Uj, Ui, Uf, Uo, UcatT);
    k_cvt_wy<<<(DIMS * HID + 255) / 256, 256, 0, stream>>>(Wy, WyT);

    // Phase 1: input projections (262144 wave jobs / 8 waves per block)
    k_input_proj<<<32768, 256, 0, stream>>>(xbf, WcatT, bj, bi, bf_, bo, pre);

    // Phase 2: recurrent scan (4 persistent WGs x 1024 threads)
    k_scan<<<4, 1024, 0, stream>>>(pre, UcatT, h0, c0, hs, hlast, clast);

    // Phase 3: output projection (32768 wave jobs / 8 per block)
    k_out_proj<<<4096, 256, 0, stream>>>(hs, WyT, by, y);
}
